// GAT_22728966930846
// MI455X (gfx1250) — matrix operand order repaired and checked by reading, not compiled, not run on hardware
//
#include <hip/hip_runtime.h>
#include <math.h>

#define NNODES   50000
#define NEDGES   800000
#define ETOT     (NNODES + NEDGES)   // 850000 incl. self loops
#define IN_CH    128
#define HID      256
#define HEADS    4
#define HEAD_DIM 64
#define OUT_CH   64
#define NEG_SLOPE 0.2f
#define BN_EPS    1e-5f

typedef __attribute__((ext_vector_type(16))) __bf16 v16bf;
typedef __attribute__((ext_vector_type(8)))  __bf16 v8bf;
typedef __attribute__((ext_vector_type(8)))  float  v8f;

#define CDIV(a,b) (((a)+(b)-1)/(b))

// ---------- monotone float<->uint encoding for atomic max ----------
__device__ __forceinline__ unsigned fenc(float f) {
    unsigned u = __float_as_uint(f);
    return (u & 0x80000000u) ? ~u : (u | 0x80000000u);
}
__device__ __forceinline__ float fdec(unsigned e) {
    unsigned u = (e & 0x80000000u) ? (e & 0x7FFFFFFFu) : ~e;
    return __uint_as_float(u);
}

// ---------- elementwise conversion / init ----------
__global__ void k_f32_to_bf16(__bf16* __restrict__ dst, const float* __restrict__ src, int n) {
    int i = blockIdx.x * blockDim.x + threadIdx.x;
    if (i < n) dst[i] = (__bf16)src[i];
}

// W [K x OC] row-major  ->  Wt [OC x K] bf16 (so B-fragment lane rows are contiguous)
__global__ void k_transpose_bf16(__bf16* __restrict__ dst, const float* __restrict__ src,
                                 int K, int OC) {
    int i = blockIdx.x * blockDim.x + threadIdx.x;
    if (i >= K * OC) return;
    int k = i / OC, oc = i % OC;
    dst[(size_t)oc * K + k] = (__bf16)src[i];
}

__global__ void k_zero_f32(float* __restrict__ p, int n) {
    int i = blockIdx.x * blockDim.x + threadIdx.x;
    if (i < n) p[i] = 0.0f;
}

// ---------- bf16 WMMA GEMM:  C[nrows x OC] = A[nrows x K] * Bt^T ----------
// One wave computes a 16x64 output tile (4 accumulators), k-step 32.
//
// A-fragment (16-bit A 16x32, contiguous k order chosen here, same as B): lane holds row M=lane%16;
//   lanes 0-15: K k0+0..15 (contiguous); lanes 16-31: K k0+16..31
//   -> base koff = hi*16, two 16B chunks at +0 and +8 elements.
// B-fragment (16-bit B 32x16, same contiguous k order as A): lane holds col N=lane%16;
//   lanes 0-15: K k0+0..15 (contiguous); lanes 16-31: K k0+16..31
//   -> base koff = hi*16, two 16B chunks at +0 and +8 elements.
__global__ void __launch_bounds__(256)
k_gemm_bf16(const __bf16* __restrict__ A, const __bf16* __restrict__ Bt,
            float* __restrict__ C, int nrows, int K, int OC) {
    const int wid  = threadIdx.x >> 5;
    const int lane = threadIdx.x & 31;
    const int rt   = blockIdx.x * 8 + wid;          // 16-row tile index
    if (rt * 16 >= nrows) return;                   // wave-uniform exit (EXEC stays full)
    const int m0   = rt * 16;
    const int n0   = blockIdx.y * 64;
    const int cl   = lane & 15;
    const int hi   = lane >> 4;                     // 0: lanes 0-15, 1: lanes 16-31

    const __bf16* ap  = A  + (size_t)(m0 + cl)      * K + hi * 16;
    const __bf16* bp0 = Bt + (size_t)(n0 +  0 + cl) * K + hi * 16;
    const __bf16* bp1 = Bt + (size_t)(n0 + 16 + cl) * K + hi * 16;
    const __bf16* bp2 = Bt + (size_t)(n0 + 32 + cl) * K + hi * 16;
    const __bf16* bp3 = Bt + (size_t)(n0 + 48 + cl) * K + hi * 16;

    v8f acc0 = {}, acc1 = {}, acc2 = {}, acc3 = {};

    for (int k0 = 0; k0 < K; k0 += 32) {
        union Frag { v16bf v; v8bf h[2]; } fa, fb0, fb1, fb2, fb3;
        fa.h[0]  = *(const v8bf*)(ap  + k0);  fa.h[1]  = *(const v8bf*)(ap  + k0 + 8);
        fb0.h[0] = *(const v8bf*)(bp0 + k0);  fb0.h[1] = *(const v8bf*)(bp0 + k0 + 8);
        fb1.h[0] = *(const v8bf*)(bp1 + k0);  fb1.h[1] = *(const v8bf*)(bp1 + k0 + 8);
        fb2.h[0] = *(const v8bf*)(bp2 + k0);  fb2.h[1] = *(const v8bf*)(bp2 + k0 + 8);
        fb3.h[0] = *(const v8bf*)(bp3 + k0);  fb3.h[1] = *(const v8bf*)(bp3 + k0 + 8);
        acc0 = __builtin_amdgcn_wmma_f32_16x16x32_bf16(false, fa.v, false, fb0.v, (short)0, acc0, false, false);
        acc1 = __builtin_amdgcn_wmma_f32_16x16x32_bf16(false, fa.v, false, fb1.v, (short)0, acc1, false, false);
        acc2 = __builtin_amdgcn_wmma_f32_16x16x32_bf16(false, fa.v, false, fb2.v, (short)0, acc2, false, false);
        acc3 = __builtin_amdgcn_wmma_f32_16x16x32_bf16(false, fa.v, false, fb3.v, (short)0, acc3, false, false);
    }

    // C/D layout: VGPR v -> row m0+v (lanes 0-15) or m0+v+8 (lanes 16-31), col = n-tile + cl
    const int rbase = m0 + hi * 8;
    for (int v = 0; v < 8; ++v) {
        float* crow = C + (size_t)(rbase + v) * OC + n0 + cl;
        crow[0]  = acc0[v];
        crow[16] = acc1[v];
        crow[32] = acc2[v];
        crow[48] = acc3[v];
    }
}

// ---------- per-node attention coefficients: a[n,h] = sum_c h[n,h,c]*att[h,c] ----------
__global__ void k_att(const float* __restrict__ h, const float* __restrict__ att,
                      float* __restrict__ out, int H, int C, int total) {
    int t = blockIdx.x * blockDim.x + threadIdx.x;
    if (t >= total) return;
    int node = t / H, hh = t % H;
    const float* hp = h + (size_t)node * H * C + (size_t)hh * C;
    const float* ap = att + hh * C;
    float s = 0.f;
    for (int c = 0; c < C; ++c) s += hp[c] * ap[c];
    out[t] = s;
}

// ---------- edge pass 1: e = leaky_relu(a_src[src]+a_dst[dst]); segment max ----------
__global__ void k_pass1(const int* __restrict__ ei, const float* __restrict__ asrc,
                        const float* __restrict__ adst, float* __restrict__ ebuf,
                        unsigned* __restrict__ menc, int H, int total) {
    int t = blockIdx.x * blockDim.x + threadIdx.x;
    if (t >= total) return;
    int et = t / H, hh = t % H;
    int src, dst;
    if (et < NEDGES) { src = ei[et]; dst = ei[NEDGES + et]; }
    else             { src = dst = et - NEDGES; }
    float e = asrc[src * H + hh] + adst[dst * H + hh];
    e = (e > 0.f) ? e : NEG_SLOPE * e;
    ebuf[t] = e;
    atomicMax(&menc[dst * H + hh], fenc(e));
}

// ---------- edge pass 2: ex = exp(e - m[dst]); segment sum ----------
__global__ void k_pass2(const int* __restrict__ ei, float* __restrict__ ebuf,
                        const unsigned* __restrict__ menc, float* __restrict__ denom,
                        int H, int total) {
    int t = blockIdx.x * blockDim.x + threadIdx.x;
    if (t >= total) return;
    int et = t / H, hh = t % H;
    int dst = (et < NEDGES) ? ei[NEDGES + et] : (et - NEDGES);
    float m = fdec(menc[dst * H + hh]);
    if (!isfinite(m)) m = 0.f;
    float ex = __expf(ebuf[t] - m);
    ebuf[t] = ex;
    atomicAdd(&denom[dst * H + hh], ex);
}

// ---------- edge pass 3: out[dst] += alpha * h[src]; one block per edge ----------
__global__ void k_scatter(const int* __restrict__ ei, const float* __restrict__ ebuf,
                          const float* __restrict__ denom, const float* __restrict__ hsrc,
                          float* __restrict__ out, int H, int HC) {
    int et = blockIdx.x;
    int c  = threadIdx.x;                 // c < HC
    int src, dst;
    if (et < NEDGES) { src = ei[et]; dst = ei[NEDGES + et]; }
    else             { src = dst = et - NEDGES; }
    int hh = c >> 6;                      // HEAD_DIM == 64 in all layers
    float alpha = ebuf[(size_t)et * H + hh] / (denom[(size_t)dst * H + hh] + 1e-16f);
    atomicAdd(&out[(size_t)dst * HC + c], alpha * hsrc[(size_t)src * HC + c]);
}

// ---------- bias + batchnorm(eval) + ELU -> bf16 for next layer ----------
__global__ void k_post(const float* __restrict__ agg, const float* __restrict__ bias,
                       const float* __restrict__ g, const float* __restrict__ be,
                       const float* __restrict__ mu, const float* __restrict__ var,
                       __bf16* __restrict__ outbf, int total) {
    int i = blockIdx.x * blockDim.x + threadIdx.x;
    if (i >= total) return;
    int j = i & (HID - 1);
    float v = agg[i] + bias[j];
    v = (v - mu[j]) * rsqrtf(var[j] + BN_EPS) * g[j] + be[j];
    v = (v > 0.f) ? v : expm1f(v);
    outbf[i] = (__bf16)v;
}

__global__ void k_bias(float* __restrict__ out, const float* __restrict__ b, int total, int oc) {
    int i = blockIdx.x * blockDim.x + threadIdx.x;
    if (i < total) out[i] += b[i % oc];
}

// =====================================================================
extern "C" void kernel_launch(void* const* d_in, const int* in_sizes, int n_in,
                              void* d_out, int out_size, void* d_ws, size_t ws_size,
                              hipStream_t stream) {
    const float* x    = (const float*)d_in[0];
    const int*   ei   = (const int*)  d_in[1];
    const float* W0   = (const float*)d_in[2];
    const float* as0  = (const float*)d_in[3];
    const float* ad0  = (const float*)d_in[4];
    const float* b0   = (const float*)d_in[5];
    const float* bn0g = (const float*)d_in[6];
    const float* bn0b = (const float*)d_in[7];
    const float* bn0m = (const float*)d_in[8];
    const float* bn0v = (const float*)d_in[9];
    const float* W1   = (const float*)d_in[10];
    const float* as1  = (const float*)d_in[11];
    const float* ad1  = (const float*)d_in[12];
    const float* b1   = (const float*)d_in[13];
    const float* bn1g = (const float*)d_in[14];
    const float* bn1b = (const float*)d_in[15];
    const float* bn1m = (const float*)d_in[16];
    const float* bn1v = (const float*)d_in[17];
    const float* W2   = (const float*)d_in[18];
    const float* as2  = (const float*)d_in[19];
    const float* ad2  = (const float*)d_in[20];
    const float* b2   = (const float*)d_in[21];
    float* out = (float*)d_out;

    // ---- workspace layout (all sizes already 256B multiples) ----
    char* ws = (char*)d_ws;
    size_t o = 0;
    __bf16*   xb    = (__bf16*)(ws + o);  o += (size_t)NNODES * IN_CH * 2;   // 12.8 MB
    __bf16*   w0t   = (__bf16*)(ws + o);  o += (size_t)HID * IN_CH * 2;
    __bf16*   w1t   = (__bf16*)(ws + o);  o += (size_t)HID * HID * 2;
    __bf16*   w2t   = (__bf16*)(ws + o);  o += (size_t)OUT_CH * HID * 2;
    float*    hA    = (float*)(ws + o);   o += (size_t)NNODES * HID * 4;     // 51.2 MB
    float*    hB    = (float*)(ws + o);   o += (size_t)NNODES * HID * 4;     // 51.2 MB
    __bf16*   hbf   = (__bf16*)(ws + o);  o += (size_t)NNODES * HID * 2;     // 25.6 MB
    float*    aS    = (float*)(ws + o);   o += (size_t)NNODES * HEADS * 4;
    float*    aD    = (float*)(ws + o);   o += (size_t)NNODES * HEADS * 4;
    unsigned* menc  = (unsigned*)(ws + o);o += (size_t)NNODES * HEADS * 4;
    float*    den   = (float*)(ws + o);   o += (size_t)NNODES * HEADS * 4;
    float*    ebuf  = (float*)(ws + o);   o += (size_t)ETOT * HEADS * 4;     // 13.6 MB

    const int ROWTILES = NNODES / 16;                 // 3125 exact
    const dim3 gemmBlk(256);
    const int gemmGX = CDIV(ROWTILES, 8);             // 391

    // ---- precompute bf16 operands ----
    k_f32_to_bf16<<<CDIV(NNODES * IN_CH, 256), 256, 0, stream>>>(xb, x, NNODES * IN_CH);
    k_transpose_bf16<<<CDIV(IN_CH * HID, 256), 256, 0, stream>>>(w0t, W0, IN_CH, HID);
    k_transpose_bf16<<<CDIV(HID * HID, 256), 256, 0, stream>>>(w1t, W1, HID, HID);
    k_transpose_bf16<<<CDIV(HID * OUT_CH, 256), 256, 0, stream>>>(w2t, W2, HID, OUT_CH);

    // =================== layer 0 (in 128 -> 4x64, concat) ===================
    k_gemm_bf16<<<dim3(gemmGX, HID / 64), gemmBlk, 0, stream>>>(xb, w0t, hA, NNODES, IN_CH, HID);
    k_att<<<CDIV(NNODES * HEADS, 256), 256, 0, stream>>>(hA, as0, aS, HEADS, HEAD_DIM, NNODES * HEADS);
    k_att<<<CDIV(NNODES * HEADS, 256), 256, 0, stream>>>(hA, ad0, aD, HEADS, HEAD_DIM, NNODES * HEADS);
    k_zero_f32<<<CDIV(NNODES * HEADS, 256), 256, 0, stream>>>((float*)menc, NNODES * HEADS);
    k_zero_f32<<<CDIV(NNODES * HEADS, 256), 256, 0, stream>>>(den, NNODES * HEADS);
    k_zero_f32<<<CDIV(NNODES * HID, 256), 256, 0, stream>>>(hB, NNODES * HID);
    k_pass1<<<CDIV(ETOT * HEADS, 256), 256, 0, stream>>>(ei, aS, aD, ebuf, menc, HEADS, ETOT * HEADS);
    k_pass2<<<CDIV(ETOT * HEADS, 256), 256, 0, stream>>>(ei, ebuf, menc, den, HEADS, ETOT * HEADS);
    k_scatter<<<ETOT, HID, 0, stream>>>(ei, ebuf, den, hA, hB, HEADS, HID);
    k_post<<<CDIV(NNODES * HID, 256), 256, 0, stream>>>(hB, b0, bn0g, bn0b, bn0m, bn0v, hbf, NNODES * HID);

    // =================== layer 1 (256 -> 4x64, concat) ===================
    k_gemm_bf16<<<dim3(gemmGX, HID / 64), gemmBlk, 0, stream>>>(hbf, w1t, hA, NNODES, HID, HID);
    k_att<<<CDIV(NNODES * HEADS, 256), 256, 0, stream>>>(hA, as1, aS, HEADS, HEAD_DIM, NNODES * HEADS);
    k_att<<<CDIV(NNODES * HEADS, 256), 256, 0, stream>>>(hA, ad1, aD, HEADS, HEAD_DIM, NNODES * HEADS);
    k_zero_f32<<<CDIV(NNODES * HEADS, 256), 256, 0, stream>>>((float*)menc, NNODES * HEADS);
    k_zero_f32<<<CDIV(NNODES * HEADS, 256), 256, 0, stream>>>(den, NNODES * HEADS);
    k_zero_f32<<<CDIV(NNODES * HID, 256), 256, 0, stream>>>(hB, NNODES * HID);
    k_pass1<<<CDIV(ETOT * HEADS, 256), 256, 0, stream>>>(ei, aS, aD, ebuf, menc, HEADS, ETOT * HEADS);
    k_pass2<<<CDIV(ETOT * HEADS, 256), 256, 0, stream>>>(ei, ebuf, menc, den, HEADS, ETOT * HEADS);
    k_scatter<<<ETOT, HID, 0, stream>>>(ei, ebuf, den, hA, hB, HEADS, HID);
    k_post<<<CDIV(NNODES * HID, 256), 256, 0, stream>>>(hB, b1, bn1g, bn1b, bn1m, bn1v, hbf, NNODES * HID);

    // =================== layer 2 (256 -> 64, heads=1, mean==identity) ===================
    k_gemm_bf16<<<dim3(gemmGX, 1), gemmBlk, 0, stream>>>(hbf, w2t, hA, NNODES, HID, OUT_CH);
    k_att<<<CDIV(NNODES, 256), 256, 0, stream>>>(hA, as2, aS, 1, OUT_CH, NNODES);
    k_att<<<CDIV(NNODES, 256), 256, 0, stream>>>(hA, ad2, aD, 1, OUT_CH, NNODES);
    k_zero_f32<<<CDIV(NNODES, 256), 256, 0, stream>>>((float*)menc, NNODES);
    k_zero_f32<<<CDIV(NNODES, 256), 256, 0, stream>>>(den, NNODES);
    k_zero_f32<<<CDIV(NNODES * OUT_CH, 256), 256, 0, stream>>>(out, NNODES * OUT_CH);
    k_pass1<<<CDIV(ETOT, 256), 256, 0, stream>>>(ei, aS, aD, ebuf, menc, 1, ETOT);
    k_pass2<<<CDIV(ETOT, 256), 256, 0, stream>>>(ei, ebuf, menc, den, 1, ETOT);
    k_scatter<<<ETOT, OUT_CH, 0, stream>>>(ei, ebuf, den, hA, out, 1, OUT_CH);
    k_bias<<<CDIV(NNODES * OUT_CH, 256), 256, 0, stream>>>(out, b2, NNODES * OUT_CH, OUT_CH);
}
